// HyperRNN_79130477462023
// MI455X (gfx1250) — matrix operand order repaired and checked by reading, not compiled, not run on hardware
//
#include <hip/hip_runtime.h>

// ---------------------------------------------------------------------------
// HyperRNN (Poincare-ball RNN) for MI455X / gfx1250, wave32 + WMMA.
//   B=256, T=128, E=H=1024, C=1 (SQC=1), EPS=1e-6
//
// Structure:
//   P0: convert W_ih, W_hh, X -> f16; zero h-state (f32 in d_out, f16 copy)
//   P1: mx = Xh @ W_ih   (WMMA GEMM, f32 accum, all-f16 operands)
//   P2: per-row prem = tanh(|mx|/|x| artanh(|x|)) mx/|mx| (f16 in-place over
//       mx, stride-2048 rows) + scalars |prem|^2, prem.b
//   per t (128x): hm = h_f16 @ W_hh (WMMA GEMM) + one fused Mobius row kernel
//       (all Mobius coefficients are analytic functions of 4 row dot-products;
//        |h|^2 carried as free scalar state).
//
// GEMM inner loop: 32x64x64 tiles, A row-major / B transposed in LDS with
// 16B-aligned 8-half runs -> ds_load_b128 fragment gathers, 2 WMMAs per tile.
// A tile staged via global_load_async_to_lds_b128 (ASYNCcnt) when available.
// ---------------------------------------------------------------------------

typedef __attribute__((ext_vector_type(16))) _Float16 v16h;
typedef __attribute__((ext_vector_type(8)))  _Float16 v8h;
typedef __attribute__((ext_vector_type(4)))  _Float16 v4h;
typedef __attribute__((ext_vector_type(2)))  _Float16 v2h;
typedef __attribute__((ext_vector_type(8)))  float    v8f;
typedef __attribute__((ext_vector_type(4)))  int      v4i;
// address-space-qualified v4i for the async-to-LDS builtin (signature learned
// from compiler diagnostic: (as1 v4i*, as3 v4i*, Ii, Ii))
typedef __attribute__((address_space(1))) v4i as1_v4i;
typedef __attribute__((address_space(3))) v4i as3_v4i;

#define EPSF 1e-6f

// ---- gfx1250 async-to-LDS path (guarded; falls back to plain load/store) ---
#if defined(__has_builtin)
#if __has_builtin(__builtin_amdgcn_global_load_async_to_lds_b128)
#define HAVE_ASYNC_LDS 1
#endif
#endif
#ifndef HAVE_ASYNC_LDS
#define HAVE_ASYNC_LDS 0
#endif

__device__ __forceinline__ void wait_asynccnt0() {
#if defined(__has_builtin)
#if __has_builtin(__builtin_amdgcn_s_wait_asynccnt)
    __builtin_amdgcn_s_wait_asynccnt(0);
    return;
#else
    asm volatile("s_wait_asynccnt 0x0" ::: "memory");
    return;
#endif
#else
    asm volatile("s_wait_asynccnt 0x0" ::: "memory");
#endif
}

__device__ __forceinline__ float artanh_clip(float x) {
    x = fminf(fmaxf(x, -1.0f + 1e-6f), 1.0f - 1e-6f);
    return 0.5f * logf((1.0f + x) / (1.0f - x));
}

// ---------------------------------------------------------------------------
// f32 -> f16 conversion, 4 elements/thread
// ---------------------------------------------------------------------------
__global__ void __launch_bounds__(256) cvt4_f16_kernel(const float* __restrict__ s,
                                                       _Float16* __restrict__ d, int n4) {
    int i = blockIdx.x * 256 + threadIdx.x;
    if (i < n4) {
        float4 v = ((const float4*)s)[i];
        v4h o;
        o[0] = (_Float16)v.x; o[1] = (_Float16)v.y;
        o[2] = (_Float16)v.z; o[3] = (_Float16)v.w;
        ((v4h*)d)[i] = o;
    }
}

// ---------------------------------------------------------------------------
// zero h-state (f32 + f16) and |h|^2
// ---------------------------------------------------------------------------
__global__ void __launch_bounds__(256) zero_kernel(float* __restrict__ h,
                                                   _Float16* __restrict__ hA,
                                                   float* __restrict__ hn2,
                                                   int nh, int nb) {
    int i = blockIdx.x * 256 + threadIdx.x;
    if (i < nh) { h[i] = 0.0f; hA[i] = (_Float16)0.0f; }
    if (i < nb) hn2[i] = 0.0f;
}

// ---------------------------------------------------------------------------
// WMMA GEMM: C[M,N](f32) = A[M,K](f16) * W[K,N](f16)
// 256 threads = 8 waves; block tile 32(M) x 64(N), K-chunks of 64 (2 WMMA).
// Ah: row-major [32][72]; Bt: transposed [n][k] = [64][72]; stride 144B keeps
// all 8-half fragment runs 16B-aligned and bank-conflict-free.
// ---------------------------------------------------------------------------
__global__ void __launch_bounds__(256) gemm_f16_wmma_kernel(
    const _Float16* __restrict__ A, const _Float16* __restrict__ W,
    float* __restrict__ C, int M, int N, int K) {
    __shared__ _Float16 Ah[32][72];
    __shared__ _Float16 Bt[64][72];

    const int tid  = threadIdx.x;
    const int wave = tid >> 5;
    const int lane = tid & 31;
    const int m    = lane & 15;
    const int lhi  = lane >> 4;            // 0: lanes 0-15, 1: lanes 16-31

    const int blockRow = blockIdx.y * 32;
    const int blockCol = blockIdx.x * 64;
    const int tRow = (wave >> 2) << 4;     // 0 or 16
    const int tCol = (wave & 3) << 4;      // 0,16,32,48

    // A staging: 32 rows x 64 halfs; one 16B (8-half) chunk per thread
    const int ar = tid >> 3, ac = (tid & 7) << 3;
    // B staging: 64(k) x 64(n); each thread: rows k0,k0+1, cols nc..nc+7,
    // scattered into Bt as packed k-pairs (ds_store_b32)
    const int k0 = (tid >> 3) << 1, nc = (tid & 7) << 3;

    v8f acc = {};

    for (int kk = 0; kk < K; kk += 64) {
        // ---- stage A (raw f16 copy; async-to-LDS when available) ----
        const _Float16* Ap = A + (size_t)(blockRow + ar) * (size_t)K + kk + ac;
#if HAVE_ASYNC_LDS
        __builtin_amdgcn_global_load_async_to_lds_b128(
            (as1_v4i*)Ap, (as3_v4i*)&Ah[ar][ac], 0, 0);
#else
        *(v8h*)&Ah[ar][ac] = *(const v8h*)Ap;
#endif
        // ---- stage B transposed ----
        const _Float16* Wp0 = W + (size_t)(kk + k0) * (size_t)N + blockCol + nc;
        const _Float16* Wp1 = Wp0 + N;
        const v8h w0 = *(const v8h*)Wp0;
        const v8h w1 = *(const v8h*)Wp1;
#pragma unroll
        for (int j = 0; j < 8; ++j) {
            v2h p; p[0] = w0[j]; p[1] = w1[j];
            *(v2h*)&Bt[nc + j][k0] = p;
        }
        if (kk + 64 < K)
            __builtin_prefetch(W + (size_t)(kk + 64 + k0) * (size_t)N + blockCol + nc, 0, 1);
#if HAVE_ASYNC_LDS
        wait_asynccnt0();
#endif
        __syncthreads();

        // ---- 2 WMMAs over the staged K=64 tile ----
#pragma unroll
        for (int kb = 0; kb < 64; kb += 32) {
            // A fragment: lane row = tRow+m; runs k = kb+8*lhi+{0..7, 16..23}
            const v8h* a0 = (const v8h*)&Ah[tRow + m][kb + (lhi << 3)];
            const v8h r0 = a0[0];
            const v8h r1 = a0[2];
            // B fragment: lane col = tCol+m; runs k = kb+8*lhi+{0..7, 16..23}
            const v8h* b0 = (const v8h*)&Bt[tCol + m][kb + (lhi << 3)];
            const v8h s0 = b0[0];
            const v8h s1 = b0[2];
            v16h afr, bfr;
#pragma unroll
            for (int e = 0; e < 8; ++e) {
                afr[e] = r0[e]; afr[8 + e] = r1[e];
                bfr[e] = s0[e]; bfr[8 + e] = s1[e];
            }
            acc = __builtin_amdgcn_wmma_f32_16x16x32_f16(false, afr, false, bfr,
                                                         (short)0, acc, false, false);
        }
        __syncthreads();
    }

    // write C: lane = column, VGPR j -> row j + 8*lhi
    const int n = blockCol + tCol + m;
#pragma unroll
    for (int j = 0; j < 8; ++j) {
        const int row = blockRow + tRow + j + (lhi << 3);
        C[(size_t)row * (size_t)N + n] = acc[j];
    }
}

// ---------------------------------------------------------------------------
// P2: per (b,t) row r of mx: prem (f16 in-place, stride-2048 rows),
// |prem|^2, prem.b. One block per row, 256 threads x 4 elems.
// ---------------------------------------------------------------------------
__global__ void __launch_bounds__(256) premix_kernel(
    const float* __restrict__ inp, float* mx, _Float16* premF16,
    const float* __restrict__ bvec, float* __restrict__ pn2, float* __restrict__ pb) {
    __shared__ float4 red[256];
    const int r = blockIdx.x, tid = threadIdx.x;
    const size_t rowOff = (size_t)r * 1024 + tid * 4;

    const float4 xv = *(const float4*)(inp + rowOff);
    const float4 mv = *(const float4*)(mx + rowOff);
    const float4 bv = *(const float4*)(bvec + tid * 4);

    float sx  = xv.x * xv.x + xv.y * xv.y + xv.z * xv.z + xv.w * xv.w;
    float sm  = mv.x * mv.x + mv.y * mv.y + mv.z * mv.z + mv.w * mv.w;
    float smb = mv.x * bv.x + mv.y * bv.y + mv.z * bv.z + mv.w * bv.w;

    red[tid] = make_float4(sx, sm, smb, 0.0f);
    __syncthreads();
    for (int o = 128; o > 0; o >>= 1) {
        if (tid < o) {
            float4 a = red[tid], b = red[tid + o];
            red[tid] = make_float4(a.x + b.x, a.y + b.y, a.z + b.z, 0.0f);
        }
        __syncthreads();
    }
    const float Sx = red[0].x, Sm = red[0].y, Smb = red[0].z;

    const float xn  = fmaxf(sqrtf(Sx), EPSF);
    const float mxn = fmaxf(sqrtf(Sm), EPSF);
    const float tau = tanhf(mxn / xn * artanh_clip(xn));
    const float tsc = tau / mxn;   // prem = tsc * mx

    // all f32 reads of this row precede the reduction barriers; safe to
    // overwrite the front half of the row with f16 prem now.
    v4h pv;
    pv[0] = (_Float16)(tsc * mv.x);
    pv[1] = (_Float16)(tsc * mv.y);
    pv[2] = (_Float16)(tsc * mv.z);
    pv[3] = (_Float16)(tsc * mv.w);
    *(v4h*)(premF16 + (size_t)r * 2048 + tid * 4) = pv;

    if (tid == 0) {
        pn2[r] = tsc * tsc * Sm;   // |prem|^2
        pb[r]  = tsc * Smb;        // prem . b
    }
}

// ---------------------------------------------------------------------------
// Fused recurrent step: one block per batch row b.
//   S1=|hm|^2, S2=hm.b, S3=prem.hm, S4=|b|^2 -> all Mobius scalars ->
//   th = tanh(cp*prem + chm*hm + cb*b) -> S5=|th|^2 ->
//   h_next = tanh(|th|)/|th| * th (f32 + f16), |h_next|^2 = f^2*S5 (free).
// ---------------------------------------------------------------------------
__global__ void __launch_bounds__(256) step_kernel(
    const float* __restrict__ hm, const _Float16* __restrict__ premF16,
    const float* __restrict__ bvec, const float* __restrict__ pn2,
    const float* __restrict__ pb, float* hstate, _Float16* hA,
    float* hn2, int t) {
    __shared__ float4 red[256];
    __shared__ float  red1[256];
    const int b = blockIdx.x, tid = threadIdx.x;
    const size_t prow = (size_t)b * 128 + t;

    const float hn2prev = hn2[b];
    const float pn2v = pn2[prow];
    const float pbv  = pb[prow];

    const float4 hv = *(const float4*)(hm + (size_t)b * 1024 + tid * 4);
    const v4h   pv4 = *(const v4h*)(premF16 + prow * 2048 + tid * 4);
    const float4 bv = *(const float4*)(bvec + tid * 4);
    float hvv[4] = {hv.x, hv.y, hv.z, hv.w};
    float pm[4]  = {(float)pv4[0], (float)pv4[1], (float)pv4[2], (float)pv4[3]};
    float bvv[4] = {bv.x, bv.y, bv.z, bv.w};

    float s1 = 0.f, s2 = 0.f, s3 = 0.f, s4 = 0.f;
#pragma unroll
    for (int i = 0; i < 4; ++i) {
        s1 += hvv[i] * hvv[i];
        s2 += hvv[i] * bvv[i];
        s3 += pm[i] * hvv[i];
        s4 += bvv[i] * bvv[i];
    }
    red[tid] = make_float4(s1, s2, s3, s4);
    __syncthreads();
    for (int o = 128; o > 0; o >>= 1) {
        if (tid < o) {
            float4 a = red[tid], c = red[tid + o];
            red[tid] = make_float4(a.x + c.x, a.y + c.y, a.z + c.z, a.w + c.w);
        }
        __syncthreads();
    }
    const float S1 = red[0].x, S2 = red[0].y, S3 = red[0].z, S4 = red[0].w;

    // mobius_matvec(h, W_hh): hidm = s * hm
    const float hn  = fmaxf(sqrtf(hn2prev), EPSF);
    const float hmn = fmaxf(sqrtf(S1), EPSF);
    const float tau = tanhf(hmn / hn * artanh_clip(hn));
    const float s   = tau / hmn;
    // mobius_add(hidm, b): hid = alpha*hidm + beta*b
    const float x2 = s * s * S1, y2 = S4, xy = s * S2;
    const float den   = fmaxf(1.f + 2.f * xy + x2 * y2, EPSF);
    const float alpha = (1.f + 2.f * xy + y2) / den;
    const float beta  = (1.f - x2) / den;
    const float hid2  = alpha * alpha * x2 + 2.f * alpha * beta * xy + beta * beta * y2;
    const float ph    = alpha * s * S3 + beta * pbv;          // prem . hid
    // mobius_add(prem, hid): z = g*prem + dd*hid
    const float den2 = fmaxf(1.f + 2.f * ph + pn2v * hid2, EPSF);
    const float g    = (1.f + 2.f * ph + hid2) / den2;
    const float dd   = (1.f - pn2v) / den2;
    const float z2   = g * g * pn2v + 2.f * g * dd * ph + dd * dd * hid2;
    // log0 scale
    const float zn = fmaxf(sqrtf(z2), EPSF);
    const float el = artanh_clip(zn) / zn;
    const float cp = el * g, chm = el * dd * alpha * s, cb = el * dd * beta;

    float th[4], s5 = 0.f;
#pragma unroll
    for (int i = 0; i < 4; ++i) {
        th[i] = tanhf(cp * pm[i] + chm * hvv[i] + cb * bvv[i]);
        s5 += th[i] * th[i];
    }
    red1[tid] = s5;
    __syncthreads();
    for (int o = 128; o > 0; o >>= 1) {
        if (tid < o) red1[tid] += red1[tid + o];
        __syncthreads();
    }
    const float S5 = red1[0];
    const float tn = fmaxf(sqrtf(S5), EPSF);
    const float f  = tanhf(tn) / tn;     // exp0 scale

    const float4 ho = make_float4(f * th[0], f * th[1], f * th[2], f * th[3]);
    *(float4*)(hstate + (size_t)b * 1024 + tid * 4) = ho;
    v4h hh;
    hh[0] = (_Float16)ho.x; hh[1] = (_Float16)ho.y;
    hh[2] = (_Float16)ho.z; hh[3] = (_Float16)ho.w;
    *(v4h*)(hA + (size_t)b * 1024 + tid * 4) = hh;
    if (tid == 0) hn2[b] = f * f * S5;
}

// ---------------------------------------------------------------------------
extern "C" void kernel_launch(void* const* d_in, const int* in_sizes, int n_in,
                              void* d_out, int out_size, void* d_ws, size_t ws_size,
                              hipStream_t stream) {
    const float* inp = (const float*)d_in[0];   // [256,128,1024] f32
    const float* Wih = (const float*)d_in[1];   // [1024,1024]    f32
    const float* Whh = (const float*)d_in[2];   // [1024,1024]    f32
    const float* bh  = (const float*)d_in[3];   // [1024]         f32
    float* hstate = (float*)d_out;              // [256,1024] f32 (h carried here)

    const int B = 256, T = 128, E = 1024, H = 1024;
    const size_t BT = (size_t)B * T;            // 32768 rows

    char* ws = (char*)d_ws;
    float*    mx      = (float*)ws;             // [32768,1024] f32 (134 MB)
    _Float16* premF16 = (_Float16*)ws;          // in-place f16, stride 2048/row
    size_t off = BT * (size_t)H * sizeof(float);
    _Float16* Xh   = (_Float16*)(ws + off); off += BT * (size_t)E * sizeof(_Float16);
    _Float16* WihH = (_Float16*)(ws + off); off += (size_t)E * H * sizeof(_Float16);
    _Float16* WhhH = (_Float16*)(ws + off); off += (size_t)H * H * sizeof(_Float16);
    float* hm  = (float*)(ws + off); off += (size_t)B * H * sizeof(float);
    _Float16* hA = (_Float16*)(ws + off); off += (size_t)B * H * sizeof(_Float16);
    float* pn2 = (float*)(ws + off); off += BT * sizeof(float);
    float* pb  = (float*)(ws + off); off += BT * sizeof(float);
    float* hn2 = (float*)(ws + off); off += (size_t)B * sizeof(float);

    // P0: conversions + state init
    cvt4_f16_kernel<<<(E * H / 4 + 255) / 256, 256, 0, stream>>>(Wih, WihH, E * H / 4);
    cvt4_f16_kernel<<<(H * H / 4 + 255) / 256, 256, 0, stream>>>(Whh, WhhH, H * H / 4);
    cvt4_f16_kernel<<<(int)((BT * E / 4 + 255) / 256), 256, 0, stream>>>(inp, Xh, (int)(BT * E / 4));
    zero_kernel<<<(B * H + 255) / 256, 256, 0, stream>>>(hstate, hA, hn2, B * H, B);

    // P1: mx = Xh @ W_ih  (one large WMMA GEMM over all timesteps)
    gemm_f16_wmma_kernel<<<dim3(H / 64, (int)(BT / 32)), 256, 0, stream>>>(
        Xh, WihH, mx, (int)BT, H, E);

    // P2: prem + per-row scalars
    premix_kernel<<<(int)BT, 256, 0, stream>>>(inp, mx, premF16, bh, pn2, pb);

    // Recurrence: 128 x (WMMA GEMM + fused Mobius row kernel)
    for (int t = 0; t < T; ++t) {
        gemm_f16_wmma_kernel<<<dim3(H / 64, B / 32), 256, 0, stream>>>(
            hA, WhhH, hm, B, H, H);
        step_kernel<<<B, 256, 0, stream>>>(hm, premF16, bh, pn2, pb, hstate, hA, hn2, t);
    }
}
